// VectorQuantizer_18631568130846
// MI455X (gfx1250) — matrix operand order repaired and checked by reading, not compiled, not run on hardware
//
#include <hip/hip_runtime.h>
#include <hip/hip_bf16.h>

typedef __bf16 bf16;
typedef __attribute__((ext_vector_type(16))) __bf16 v16bf;
typedef __attribute__((ext_vector_type(8)))  __bf16 v8bf;
typedef __attribute__((ext_vector_type(8)))  float  v8f;

// exact parameter types for __builtin_amdgcn_global_load_async_to_lds_b128
typedef int v4i __attribute__((vector_size(16)));
typedef __attribute__((address_space(1))) v4i* as1_v4i_ptr;   // global
typedef __attribute__((address_space(3))) v4i* as3_v4i_ptr;   // LDS

#define K_CODES 1024
#define DIM     256
#define N_ROWS  65536
#define BM      128              // rows per workgroup (8 waves x 16)
#define CHUNK   32               // codewords per LDS buffer (double-buffered)
#define NCHUNK  (K_CODES / CHUNK)
#define BPAD    8                // bf16 padding per LDS codeword row
#define BSTRIDE (DIM + BPAD)     // 264 bf16 = 528 B = 132 dwords == 4 mod 64 banks
#define QELEMS  (N_ROWS * DIM)   // 16777216 floats of quantized output

#if defined(__gfx1250__) && __has_builtin(__builtin_amdgcn_global_load_async_to_lds_b128)
#define HAS_ASYNC_LDS 1
#else
#define HAS_ASYNC_LDS 0
#endif

// Issue one chunk's codebook copy (16 KB) into a padded LDS buffer.
// Async path: zero VGPR cost, tracked by ASYNCcnt. Fallback: reg copy.
__device__ __forceinline__ void load_chunk(const bf16* __restrict__ cb,
                                           bf16* dstB, int c, int tid) {
    const uint4* g = (const uint4*)(cb + (size_t)c * CHUNK * DIM);
    uint4*       d = (uint4*)dstB;
#pragma unroll
    for (int k = 0; k < 4; ++k) {
        const int gi = tid + k * 256;               // 32 uint4 per source row
        const int li = (gi >> 5) * 33 + (gi & 31);  // padded row = 33 uint4
#if HAS_ASYNC_LDS
        __builtin_amdgcn_global_load_async_to_lds_b128(
            (as1_v4i_ptr)(g + gi), (as3_v4i_ptr)(d + li), 0, 0);
#else
        d[li] = g[gi];
#endif
    }
}

__device__ __forceinline__ void wait_async_done() {
#if HAS_ASYNC_LDS
#if __has_builtin(__builtin_amdgcn_s_wait_asynccnt)
    __builtin_amdgcn_s_wait_asynccnt(0);
#else
    asm volatile("s_wait_asynccnt 0x0" ::: "memory");
#endif
#endif
}

// load one 32x16 bf16 B tile slice (dims 8h+0..7 and 16+8h+0..7 of one codeword/lane)
__device__ __forceinline__ v16bf loadB(const bf16* base, int s) {
    const v8bf lo = *(const v8bf*)(base + s * 32);
    const v8bf hi = *(const v8bf*)(base + s * 32 + 16);
    return __builtin_shufflevector(lo, hi, 0, 1, 2, 3, 4, 5, 6, 7,
                                   8, 9, 10, 11, 12, 13, 14, 15);
}

// ---------------------------------------------------------------------------
// Kernel 1: codebook -> bf16 copy in workspace, plus per-codeword ||e||^2
// ---------------------------------------------------------------------------
__global__ __launch_bounds__(64) void vq_prep(const float* __restrict__ w,
                                              bf16* __restrict__ cb,
                                              float* __restrict__ norms) {
    __shared__ float red[64];
    const int k = blockIdx.x;
    const int t = threadIdx.x;
    const float4 v = ((const float4*)w)[k * 64 + t];
    float sq = v.x * v.x + v.y * v.y + v.z * v.z + v.w * v.w;
    const int o = k * DIM + t * 4;
    cb[o + 0] = (bf16)v.x; cb[o + 1] = (bf16)v.y;
    cb[o + 2] = (bf16)v.z; cb[o + 3] = (bf16)v.w;
    red[t] = sq;
    __syncthreads();
    for (int s = 32; s > 0; s >>= 1) {
        if (t < s) red[t] += red[t + s];
        __syncthreads();
    }
    if (t == 0) norms[k] = red[0];
}

// ---------------------------------------------------------------------------
// Kernel 2: fused  (-2 x.e + ||e||^2)  GEMM + running argmin + loss partials.
// A strip (16 rows x 256 dims, bf16) lives in registers per wave; codebook
// double-buffered in LDS with async global->LDS prefetch of the next chunk;
// B tiles software-pipelined one K-step ahead of the WMMAs.
// ---------------------------------------------------------------------------
__global__ __launch_bounds__(256) void vq_main(const float* __restrict__ x,
                                               const bf16* __restrict__ cb,
                                               const float* __restrict__ norms,
                                               float* __restrict__ idxf_out,
                                               float* __restrict__ partial) {
    __shared__ bf16  sB[2][CHUNK * BSTRIDE];  // 2 x 16.5 KB padded chunks
    __shared__ float sNorm[K_CODES];          // 4 KB  ||e||^2
    __shared__ float sRowNorm[BM];            // ||x||^2 per row
    __shared__ float sLoss[16];

    const int tid  = threadIdx.x;
    const int wave = tid >> 5;
    const int lane = tid & 31;
    const int r    = lane & 15;   // row within strip (A) / column within tile (B,D)
    const int h    = lane >> 4;   // half-wave selector

    // kick off chunk 0 prefetch immediately (overlaps with A-strip load below)
    load_chunk(cb, &sB[0][0], 0, tid);

    for (int i = tid; i < K_CODES; i += 256) sNorm[i] = norms[i];

    // ---- load A strip (f32 -> bf16) into registers, accumulate ||x||^2 ----
    const int    row_g = blockIdx.x * BM + wave * 16 + r;
    const float* xr    = x + row_g * DIM;
    v16bf A[8];
    float sq = 0.f;
#pragma unroll
    for (int s = 0; s < 8; ++s) {
        // lane<16 : elements 0..7 = dims s*32+0..7,  8..15 = dims s*32+16..23
        // lane>=16: elements 0..7 = dims s*32+8..15, 8..15 = dims s*32+24..31
        const float4 f0 = *(const float4*)(xr + s * 32 + h * 8);
        const float4 f1 = *(const float4*)(xr + s * 32 + h * 8 + 4);
        const float4 f2 = *(const float4*)(xr + s * 32 + 16 + h * 8);
        const float4 f3 = *(const float4*)(xr + s * 32 + 16 + h * 8 + 4);
        A[s][0]  = (bf16)f0.x; A[s][1]  = (bf16)f0.y; A[s][2]  = (bf16)f0.z; A[s][3]  = (bf16)f0.w;
        A[s][4]  = (bf16)f1.x; A[s][5]  = (bf16)f1.y; A[s][6]  = (bf16)f1.z; A[s][7]  = (bf16)f1.w;
        A[s][8]  = (bf16)f2.x; A[s][9]  = (bf16)f2.y; A[s][10] = (bf16)f2.z; A[s][11] = (bf16)f2.w;
        A[s][12] = (bf16)f3.x; A[s][13] = (bf16)f3.y; A[s][14] = (bf16)f3.z; A[s][15] = (bf16)f3.w;
        sq += f0.x * f0.x + f0.y * f0.y + f0.z * f0.z + f0.w * f0.w;
        sq += f1.x * f1.x + f1.y * f1.y + f1.z * f1.z + f1.w * f1.w;
        sq += f2.x * f2.x + f2.y * f2.y + f2.z * f2.z + f2.w * f2.w;
        sq += f3.x * f3.x + f3.y * f3.y + f3.z * f3.z + f3.w * f3.w;
    }
    sq += __shfl_xor(sq, 16, 32);             // combine the two half-rows
    if (h == 0) sRowNorm[wave * 16 + r] = sq;

    float minv[8];
    int   mini[8];
#pragma unroll
    for (int j = 0; j < 8; ++j) { minv[j] = 3.4e38f; mini[j] = 0; }

    wait_async_done();
    __syncthreads();                          // chunk 0 + norms + row norms ready

    for (int c = 0; c < NCHUNK; ++c) {
        const int cur = c & 1;
        if (c + 1 < NCHUNK)                   // async prefetch of next chunk
            load_chunk(cb, &sB[cur ^ 1][0], c + 1, tid);

        const bf16* bp0 = &sB[cur][0] + (0 * 16 + r) * BSTRIDE + h * 8;
        const bf16* bp1 = &sB[cur][0] + (1 * 16 + r) * BSTRIDE + h * 8;

        v8f acc0 = (v8f){0.f,0.f,0.f,0.f,0.f,0.f,0.f,0.f};
        v8f acc1 = (v8f){0.f,0.f,0.f,0.f,0.f,0.f,0.f,0.f};

        // software pipeline: B tiles loaded one K-step ahead of the WMMAs
        v16bf b0 = loadB(bp0, 0);
        v16bf b1 = loadB(bp1, 0);
#pragma unroll
        for (int s = 0; s < 8; ++s) {
            v16bf n0 = b0, n1 = b1;
            if (s < 7) { n0 = loadB(bp0, s + 1); n1 = loadB(bp1, s + 1); }
            acc0 = __builtin_amdgcn_wmma_f32_16x16x32_bf16(
                false, A[s], false, b0, (short)0, acc0, false, false);
            acc1 = __builtin_amdgcn_wmma_f32_16x16x32_bf16(
                false, A[s], false, b1, (short)0, acc1, false, false);
            b0 = n0; b1 = n1;
        }

        {
            const int   col0 = c * CHUNK + r;
            const int   col1 = col0 + 16;
            const float en0  = sNorm[col0];
            const float en1  = sNorm[col1];
#pragma unroll
            for (int j = 0; j < 8; ++j) {     // acc[j]: row j (lanes<16) / j+8
                const float d0 = fmaf(-2.f, acc0[j], en0);
                if (d0 < minv[j]) { minv[j] = d0; mini[j] = col0; }
                const float d1 = fmaf(-2.f, acc1[j], en1);
                if (d1 < minv[j]) { minv[j] = d1; mini[j] = col1; }
            }
        }

        wait_async_done();                    // own async stores to LDS complete
        __syncthreads();                      // all waves' prefetch visible
    }

    // ---- cross-lane argmin within each 16-lane column group ----
#pragma unroll
    for (int m = 1; m < 16; m <<= 1) {
#pragma unroll
        for (int j = 0; j < 8; ++j) {
            const float ov = __shfl_xor(minv[j], m, 32);
            const int   oi = __shfl_xor(mini[j], m, 32);
            if (ov < minv[j] || (ov == minv[j] && oi < mini[j])) {
                minv[j] = ov; mini[j] = oi;
            }
        }
    }

    if (r == 0) {   // lane 0 -> rows 0..7 of strip, lane 16 -> rows 8..15
        const int rbase = blockIdx.x * BM + wave * 16 + h * 8;
        float lsum = 0.f;
#pragma unroll
        for (int j = 0; j < 8; ++j) {
            const int row = rbase + j;
            idxf_out[row] = (float)mini[j];
            lsum += minv[j] + sRowNorm[wave * 16 + h * 8 + j];  // ||x - e_best||^2
        }
        sLoss[wave * 2 + h] = lsum;
    }
    __syncthreads();
    if (tid == 0) {
        float s = 0.f;
#pragma unroll
        for (int i = 0; i < 16; ++i) s += sLoss[i];
        partial[blockIdx.x] = s;   // deterministic: one writer per slot
    }
}

// ---------------------------------------------------------------------------
// Kernel 3: reduce 512 workgroup partials -> final scalar loss
// ---------------------------------------------------------------------------
__global__ __launch_bounds__(512) void vq_loss(const float* __restrict__ partial,
                                               float* __restrict__ loss_out) {
    __shared__ float red[512];
    const int t = threadIdx.x;
    red[t] = partial[t];
    __syncthreads();
    for (int s = 256; s > 0; s >>= 1) {
        if (t < s) red[t] += red[t + s];
        __syncthreads();
    }
    if (t == 0) loss_out[0] = 1.25f * red[0] / (float)QELEMS;
}

// ---------------------------------------------------------------------------
// Kernel 4: quantized_st[row] = weight[idx[row]]  (float4 gather/stores)
// ---------------------------------------------------------------------------
__global__ __launch_bounds__(256) void vq_gather(const float* __restrict__ w,
                                                 const float* __restrict__ idxf,
                                                 float* __restrict__ q) {
    const int t   = threadIdx.x;
    const int row = blockIdx.x * 4 + (t >> 6);
    const int d4  = t & 63;
    const int id  = (int)idxf[row];
    ((float4*)q)[row * 64 + d4] = ((const float4*)w)[id * 64 + d4];
}

// ---------------------------------------------------------------------------
extern "C" void kernel_launch(void* const* d_in, const int* in_sizes, int n_in,
                              void* d_out, int out_size, void* d_ws, size_t ws_size,
                              hipStream_t stream) {
    (void)in_sizes; (void)n_in; (void)out_size; (void)ws_size;
    const float* x = (const float*)d_in[0];   // [16,4096,256] f32
    const float* w = (const float*)d_in[1];   // [1024,256]    f32

    float* out  = (float*)d_out;
    float* q    = out;                 // [N, D] quantized_st
    float* loss = out + QELEMS;        // scalar
    float* idxf = out + QELEMS + 1;    // [N] indices as float

    char*  ws      = (char*)d_ws;
    bf16*  cb      = (bf16*)ws;                      // 512 KB bf16 codebook
    float* norms   = (float*)(ws + 524288);          // 4 KB  ||e||^2
    float* partial = (float*)(ws + 528384);          // 2 KB  loss partials

    vq_prep  <<<K_CODES, 64, 0, stream>>>(w, cb, norms);
    vq_main  <<<N_ROWS / BM, 256, 0, stream>>>(x, cb, norms, idxf, partial);
    vq_loss  <<<1, 512, 0, stream>>>(partial, loss);
    vq_gather<<<N_ROWS / 4, 256, 0, stream>>>(w, idxf, q);
}
